// GRUQNetwork_68101001445662
// MI455X (gfx1250) — compile-verified, matrix operand order checked by reading, not run on hardware
//
#include <hip/hip_runtime.h>
#include <stdint.h>
#include <stddef.h>

// GRU Q-network forward for MI455X (gfx1250), wave32.
//   Phase 0: one-time f32->bf16 conversion of x and all GEMM weights (bandwidth-trivial).
//   Phase 1: xp     = mish(x @ W_pre^T + b_pre)          bf16 WMMA, M=8192 N=1024 K=256
//   Phase 2: igates = xp @ w_ih^T + b_ih                 bf16 WMMA, M=8192 N=3072 K=1024
//   Phase 3: sequential GRU scan, weight-stationary w_hh slices in LDS across 64 WGPs,
//            per-step 4KB h exchange via L2 + grid barrier (fp32 recurrence).
//   Phase 4: y1,y2 post MLPs                             bf16 WMMA
//   Phase 5: dueling heads + L2-normalized advantage, one wave per timestep.
// GEMM: 64x64 per wave (4x4 WMMA block, accumulators resident; __launch_bounds__(256,1)
// lifts the VGPR cap). K iterated in 128-element macro-steps; inside, 4 k-steps use
// compile-time IOFFSETs so the loop body is global_load_b128(+imm) -> v_wmma with only
// 8 pointer increments per macro-step.

#define T_LEN 8192
#define OBS   256
#define ACT   16
#define MLP   1024
#define REC   1024

typedef __attribute__((ext_vector_type(16))) __bf16 v16bf;
typedef __attribute__((ext_vector_type(8)))  __bf16 v8bf;
typedef __attribute__((ext_vector_type(4)))  __bf16 v4bf;
typedef __attribute__((ext_vector_type(8)))  float  v8f;
typedef __attribute__((ext_vector_type(4)))  float  v4f;

// global (address_space(1)) pointer aliases
typedef const __attribute__((address_space(1))) v8bf*   g_v8bf_cp;
typedef const __attribute__((address_space(1))) v4f*    g_v4f_cp;
typedef const __attribute__((address_space(1))) __bf16* g_bf_cp;
typedef const __attribute__((address_space(1))) float*  g_f32_cp;
typedef __attribute__((address_space(1))) __bf16* g_bf_p;
typedef __attribute__((address_space(1))) float*  g_f32_p;
typedef __attribute__((address_space(1))) v4bf*   g_v4bf_p;

// ---------- scalar helpers ----------
__device__ __forceinline__ __bf16 f2bf(float f) {
  union { float f; uint32_t u; } c; c.f = f;
  uint32_t u = c.u;
  u += 0x7FFFu + ((u >> 16) & 1u);          // round-to-nearest-even
  union { uint16_t s; __bf16 b; } r; r.s = (uint16_t)(u >> 16);
  return r.b;
}
__device__ __forceinline__ float bf2f(__bf16 b) {
  union { __bf16 b; uint16_t s; } c; c.b = b;
  union { uint32_t u; float f; } r; r.u = ((uint32_t)c.s) << 16;
  return r.f;
}
__device__ __forceinline__ float mishf(float v) {
  float sp = (v > 20.0f) ? v : logf(1.0f + expf(v));   // softplus
  return v * tanhf(sp);
}
__device__ __forceinline__ float sigmoidf(float x) {
  return 1.0f / (1.0f + expf(-x));
}

// ---------- one-time f32 -> bf16 tensor conversion (vectorized 4-wide) ----------
__global__ __launch_bounds__(256) void cvt_f32_bf16(const float* __restrict__ in,
                                                    __bf16* __restrict__ out, int n) {
  int i = (blockIdx.x * 256 + threadIdx.x) * 4;
  if (i >= n) return;
  v4f x = *(g_v4f_cp)(in + i);
  v4bf o;
#pragma unroll
  for (int j = 0; j < 4; ++j) o[j] = f2bf(x[j]);
  *(g_v4bf_p)(out + i) = o;
}

// ---------- WMMA fragment load (ISA 7.12.2 layouts) ----------
// A (16x32, 16-bit): lanes 0-15: row M=lane, K = k0+{0..7} in v0..v3, k0+{16..23} in v4..v7
//                    lanes 16-31: row M=lane-16, same K pattern shifted by +8.
// B (32x16, 16-bit) of op(B)=W^T (W row-major [N,K]): lane%16 = column N, same K mapping.
// e8 is a compile-time offset in v8bf (16-byte) units -> immediate-offset global_load_b128.
__device__ __forceinline__ v16bf load_frag_o(g_v8bf_cp gp, int e8) {
  union { v16bf v; v8bf h[2]; } u;
  u.h[0] = gp[e8];        // K run k..k+7
  u.h[1] = gp[e8 + 2];    // K run k+16..k+23 (+32 bytes)
  return u.v;
}
__device__ __forceinline__ v8f wmma_bf16(v16bf a, v16bf b, v8f c) {
  return __builtin_amdgcn_wmma_f32_16x16x32_bf16(false, a, false, b, (short)0, c,
                                                 false, false);
}

// C/D layout: lane%16 = N; VGPR r -> M = r + 8*(lane/16).
template <int MODE>  // 0: +bias, store f32.  1: +bias, mish, store bf16.
__device__ __forceinline__ void store_tile(void* __restrict__ outp, int ldc, int m0, int n0,
                                           v8f c, const float* __restrict__ bias, int lane) {
  int nn = n0 + (lane & 15);
  int mb = m0 + ((lane >> 4) << 3);
  float bn = ((g_f32_cp)bias)[nn];
#pragma unroll
  for (int r = 0; r < 8; ++r) {
    float v = c[r] + bn;
    size_t idx = (size_t)(mb + r) * ldc + nn;
    if (MODE == 1) ((g_bf_p)outp)[idx] = f2bf(mishf(v));
    else           ((g_f32_p)outp)[idx] = v;
  }
}

// ---------- GEMM: C = op(A@W^T + b). 8 waves/block, 64x64 per wave (4x4 frag block) ----------
template <int MODE, int K>
__global__ __launch_bounds__(256, 1)
void gemm_bf16(const __bf16* __restrict__ A, const __bf16* __restrict__ W,
               const float* __restrict__ bias, void* __restrict__ outp,
               int M, int N) {
  int lane = threadIdx.x & 31;
  int id   = blockIdx.x * 8 + (threadIdx.x >> 5);
  int tN   = N >> 6;
  int m0   = (id / tN) << 6;
  int n0   = (id % tN) << 6;
  if (m0 >= M) return;

  int rsub = lane & 15;               // row-in-tile
  int koff = (lane >> 4) << 3;        // +8 K for upper half-lanes
  g_v8bf_cp pa[4], pb[4];
#pragma unroll
  for (int i = 0; i < 4; ++i) {
    pa[i] = (g_v8bf_cp)(A + (size_t)(m0 + 16 * i + rsub) * K + koff);
    pb[i] = (g_v8bf_cp)(W + (size_t)(n0 + 16 * i + rsub) * K + koff);
  }

  v8f acc[4][4] = {};
  for (int k0 = 0; k0 < K; k0 += 128) {      // macro-step: 4 k-steps of 32
#pragma unroll
    for (int kk = 0; kk < 4; ++kk) {         // compile-time IOFFSETs 0/64/128/192 B
      v16bf fa[4], fb[4];
#pragma unroll
      for (int i = 0; i < 4; ++i) {
        fa[i] = load_frag_o(pa[i], kk * 4);
        fb[i] = load_frag_o(pb[i], kk * 4);
      }
#pragma unroll
      for (int i = 0; i < 4; ++i)
#pragma unroll
        for (int j = 0; j < 4; ++j)
          acc[i][j] = wmma_bf16(fa[i], fb[j], acc[i][j]);
    }
#pragma unroll
    for (int i = 0; i < 4; ++i) { pa[i] += 16; pb[i] += 16; }  // +128 elements
  }
#pragma unroll
  for (int i = 0; i < 4; ++i)
#pragma unroll
    for (int j = 0; j < 4; ++j)
      store_tile<MODE>(outp, N, m0 + 16 * i, n0 + 16 * j, acc[i][j], bias, lane);
}

// ---------- grid barrier (generation counter, device scope, s_sleep spin) ----------
__device__ __forceinline__ void grid_barrier(int* bar, int nblocks, int tid) {
  __syncthreads();
  if (tid == 0) {
    int gen = __hip_atomic_load(&bar[1], __ATOMIC_ACQUIRE, __HIP_MEMORY_SCOPE_AGENT);
    int arrived = __hip_atomic_fetch_add(&bar[0], 1, __ATOMIC_ACQ_REL, __HIP_MEMORY_SCOPE_AGENT);
    if (arrived == nblocks - 1) {
      __hip_atomic_store(&bar[0], 0, __ATOMIC_RELAXED, __HIP_MEMORY_SCOPE_AGENT);
      __hip_atomic_fetch_add(&bar[1], 1, __ATOMIC_RELEASE, __HIP_MEMORY_SCOPE_AGENT);
    } else {
      while (__hip_atomic_load(&bar[1], __ATOMIC_ACQUIRE, __HIP_MEMORY_SCOPE_AGENT) == gen)
        __builtin_amdgcn_s_sleep(2);
    }
  }
  __syncthreads();
}

// ---------- persistent GRU scan: 64 blocks, each owns 16 h-indices ----------
// LDS: w_hh slice [48][1024] f32 (192 KB) + h stage [1024] + hg [48]  -> dynamic LDS (~197 KB/WGP).
__global__ __launch_bounds__(256)
void gru_scan(const float* __restrict__ igates, const float* __restrict__ w_hh,
              const float* __restrict__ b_n, const float* __restrict__ state0,
              const unsigned char* __restrict__ start,
              float* __restrict__ h_ws, __bf16* __restrict__ states,
              int* __restrict__ bar) {
  extern __shared__ float smem[];
  float* Wl = smem;               // [48][REC]
  float* hs = smem + 48 * REC;    // [REC]
  float* hg = hs + REC;           // [48]
  const int b = blockIdx.x, tid = threadIdx.x;
  const int lane = tid & 31, wave = tid >> 5;

  // Stage this block's 48 w_hh rows (r/z/n row triples for owned h indices) into LDS.
  for (int j = wave; j < 48; j += 8) {
    int grow = (j < 16) ? (b * 16 + j)
             : (j < 32) ? (REC + b * 16 + (j - 16))
                        : (2 * REC + b * 16 + (j - 32));
    g_f32_cp src = (g_f32_cp)(w_hh + (size_t)grow * REC);
    for (int c = lane; c < REC; c += 32) Wl[j * REC + c] = src[c];
  }
  if (b == 0)
    for (int i = tid; i < REC; i += 256) h_ws[i] = state0[i];
  __syncthreads();
  __threadfence();
  grid_barrier(bar, (int)gridDim.x, tid);

  for (int t = 0; t < T_LEN; ++t) {
    float ns = start[t] ? 0.0f : 1.0f;                 // reset at episode starts
    for (int i = tid; i < REC; i += 256) hs[i] = ((g_f32_cp)h_ws)[i] * ns;
    __syncthreads();

    // 6 rows per wave: lane-strided dot over LDS (conflict-free), shfl_xor reduce.
#pragma unroll
    for (int jj = 0; jj < 6; ++jj) {
      int j = wave * 6 + jj;
      const float* wrow = Wl + j * REC;
      float s = 0.0f;
#pragma unroll
      for (int k = 0; k < REC / 32; ++k)
        s = fmaf(wrow[lane + 32 * k], hs[lane + 32 * k], s);
#pragma unroll
      for (int o = 16; o > 0; o >>= 1) s += __shfl_xor(s, o, 32);
      if (lane == 0) hg[j] = s;
    }
    __syncthreads();

    if (tid < 16) {
      int gi = b * 16 + tid;
      g_f32_cp ig = (g_f32_cp)(igates + (size_t)t * (3 * REC));
      float r = sigmoidf(ig[gi]           + hg[tid]);
      float z = sigmoidf(ig[REC + gi]     + hg[16 + tid]);
      float n = tanhf   (ig[2 * REC + gi] + r * (hg[32 + tid] + b_n[gi]));
      float ho = hs[gi];                      // already reset-scaled
      float hn = n + z * (ho - n);            // equinox GRUCell convention
      ((g_f32_p)h_ws)[gi] = hn;
      ((g_bf_p)states)[(size_t)t * REC + gi] = f2bf(hn);
    }
    __syncthreads();
    __threadfence();
    grid_barrier(bar, (int)gridDim.x, tid);
  }
}

// ---------- heads: one wave per timestep ----------
__global__ __launch_bounds__(256)
void heads_kernel(const __bf16* __restrict__ y2,
                  const float* __restrict__ Wv, const float* __restrict__ bv,
                  const float* __restrict__ Wa, const float* __restrict__ ba,
                  const float* __restrict__ Wsc, const float* __restrict__ bsc,
                  const float* __restrict__ h_final, float* __restrict__ out) {
  int lane = threadIdx.x & 31, wave = threadIdx.x >> 5;
  int t = blockIdx.x * 8 + wave;
  if (blockIdx.x == 0)  // append final_state after q in d_out
    for (int i = threadIdx.x; i < REC; i += 256)
      ((g_f32_p)out)[(size_t)T_LEN * ACT + i] = ((g_f32_cp)h_final)[i];
  if (t >= T_LEN) return;

  g_bf_cp y = (g_bf_cp)(y2 + (size_t)t * MLP);
  float av = 0.0f, as = 0.0f, aa[ACT];
#pragma unroll
  for (int j = 0; j < ACT; ++j) aa[j] = 0.0f;
  for (int k = 0; k < MLP / 32; ++k) {
    int c = lane + 32 * k;
    float yv = bf2f(y[c]);
    av = fmaf(yv, ((g_f32_cp)Wv)[c], av);
    as = fmaf(yv, ((g_f32_cp)Wsc)[c], as);
#pragma unroll
    for (int j = 0; j < ACT; ++j) aa[j] = fmaf(yv, ((g_f32_cp)Wa)[j * MLP + c], aa[j]);
  }
#pragma unroll
  for (int o = 16; o > 0; o >>= 1) {
    av += __shfl_xor(av, o, 32);
    as += __shfl_xor(as, o, 32);
#pragma unroll
    for (int j = 0; j < ACT; ++j) aa[j] += __shfl_xor(aa[j], o, 32);
  }
  if (lane == 0) {
    float v = av + bv[0], s = as + bsc[0];
    float a[ACT], n2 = 0.0f;
#pragma unroll
    for (int j = 0; j < ACT; ++j) { a[j] = aa[j] + ba[j]; n2 += a[j] * a[j]; }
    float inv = 1.0f / (1e-6f + sqrtf(n2));
    float mean = 0.0f;
#pragma unroll
    for (int j = 0; j < ACT; ++j) { a[j] *= inv; mean += a[j]; }
    mean *= (1.0f / ACT);
#pragma unroll
    for (int j = 0; j < ACT; ++j)
      ((g_f32_p)out)[(size_t)t * ACT + j] = v + s * (a[j] - mean);
  }
}

// ---------- host launch ----------
extern "C" void kernel_launch(void* const* d_in, const int* in_sizes, int n_in,
                              void* d_out, int out_size, void* d_ws, size_t ws_size,
                              hipStream_t stream) {
  (void)in_sizes; (void)n_in; (void)out_size; (void)ws_size;
  const float*         x     = (const float*)d_in[0];
  const float*         state = (const float*)d_in[1];
  const unsigned char* start = (const unsigned char*)d_in[2];  // jnp bool_ = 1 byte
  const float* W_pre = (const float*)d_in[3];
  const float* b_pre = (const float*)d_in[4];
  const float* w_ih  = (const float*)d_in[5];
  const float* w_hh  = (const float*)d_in[6];
  const float* b_ih  = (const float*)d_in[7];
  const float* b_n   = (const float*)d_in[8];
  const float* W1    = (const float*)d_in[9];
  const float* b1    = (const float*)d_in[10];
  const float* W2    = (const float*)d_in[11];
  const float* b2    = (const float*)d_in[12];
  const float* Wv    = (const float*)d_in[13];
  const float* bv    = (const float*)d_in[14];
  const float* Wa    = (const float*)d_in[15];
  const float* ba    = (const float*)d_in[16];
  const float* Wsc   = (const float*)d_in[17];
  const float* bsc   = (const float*)d_in[18];

  char* ws = (char*)d_ws;
  size_t off = 0;
  __bf16* xp     = (__bf16*)(ws + off); off += (size_t)T_LEN * MLP * 2;       // 16 MB
  float*  igates = (float*) (ws + off); off += (size_t)T_LEN * 3 * REC * 4;   // 96 MB
  __bf16* statesb= (__bf16*)(ws + off); off += (size_t)T_LEN * REC * 2;       // 16 MB
  __bf16* y1p    = (__bf16*)(ws + off); off += (size_t)T_LEN * MLP * 2;       // 16 MB
  __bf16* y2p    = (__bf16*)(ws + off); off += (size_t)T_LEN * MLP * 2;       // 16 MB
  __bf16* xb     = (__bf16*)(ws + off); off += (size_t)T_LEN * OBS * 2;       // 4 MB
  __bf16* wpre_b = (__bf16*)(ws + off); off += (size_t)MLP * OBS * 2;         // 0.5 MB
  __bf16* wih_b  = (__bf16*)(ws + off); off += (size_t)3 * REC * MLP * 2;     // 6 MB
  __bf16* w1_b   = (__bf16*)(ws + off); off += (size_t)MLP * REC * 2;         // 2 MB
  __bf16* w2_b   = (__bf16*)(ws + off); off += (size_t)MLP * MLP * 2;         // 2 MB
  float*  h_ws   = (float*) (ws + off); off += (size_t)REC * 4;
  int*    bar    = (int*)   (ws + off); off += 2 * sizeof(int);

  hipMemsetAsync(bar, 0, 2 * sizeof(int), stream);

  // Phase 0: one-time bf16 conversions (elementwise, 4-wide).
  {
    struct { const float* src; __bf16* dst; int n; } cv[5] = {
      { x,     xb,     T_LEN * OBS },
      { W_pre, wpre_b, MLP * OBS },
      { w_ih,  wih_b,  3 * REC * MLP },
      { W1,    w1_b,   MLP * REC },
      { W2,    w2_b,   MLP * MLP },
    };
    for (int i = 0; i < 5; ++i)
      cvt_f32_bf16<<<(cv[i].n / 4 + 255) / 256, 256, 0, stream>>>(cv[i].src, cv[i].dst, cv[i].n);
  }

  // Phase 1: xp = mish(x @ W_pre^T + b_pre)
  gemm_bf16<1, OBS><<<(T_LEN / 64) * (MLP / 64) / 8, 256, 0, stream>>>(
      xb, wpre_b, b_pre, (void*)xp, T_LEN, MLP);
  // Phase 2: igates = xp @ w_ih^T + b_ih   (f32 out)
  gemm_bf16<0, MLP><<<(T_LEN / 64) * (3 * REC / 64) / 8, 256, 0, stream>>>(
      xp, wih_b, b_ih, (void*)igates, T_LEN, 3 * REC);
  // Phase 3: persistent weight-stationary scan, 64 WGPs
  size_t smem = (size_t)(48 * REC + REC + 64) * sizeof(float);
  hipFuncSetAttribute((const void*)gru_scan,
                      hipFuncAttributeMaxDynamicSharedMemorySize, (int)smem);
  gru_scan<<<64, 256, smem, stream>>>(igates, w_hh, b_n, state, start,
                                      h_ws, statesb, bar);
  // Phase 4: post MLPs (bf16 out + mish)
  gemm_bf16<1, REC><<<(T_LEN / 64) * (MLP / 64) / 8, 256, 0, stream>>>(
      statesb, w1_b, b1, (void*)y1p, T_LEN, MLP);
  gemm_bf16<1, MLP><<<(T_LEN / 64) * (MLP / 64) / 8, 256, 0, stream>>>(
      y1p, w2_b, b2, (void*)y2p, T_LEN, MLP);
  // Phase 5: heads + dueling combine; also appends final_state to d_out
  heads_kernel<<<T_LEN / 8, 256, 0, stream>>>(y2p, Wv, bv, Wa, ba, Wsc, bsc,
                                              h_ws, (float*)d_out);
}
